// CorrBlock1D_21268678050371
// MI455X (gfx1250) — matrix operand order repaired and checked by reading, not compiled, not run on hardware
//
#include <hip/hip_runtime.h>

// Problem constants (from setup_inputs): B=4, C=256, H=80, W1=W2=184, G=2, S=9, L=4
#define BB   4
#define CCH  256
#define HH   80
#define WW1  184
#define WW2  184
#define NG   2
#define NS   9
#define NL   4

typedef __attribute__((ext_vector_type(16))) _Float16    v16h;
typedef __attribute__((ext_vector_type(8)))  float       v8f;
typedef __attribute__((ext_vector_type(4)))  unsigned int v4u;

union FragU { v16h h; v4u u[2]; };
union PackU { _Float16 h[2]; unsigned int u; };

// ---------------------------------------------------------------------------
// Kernel 1: all-pairs 1D correlation via f16 WMMA (fp32 accumulate).
// For each (b,h): corr[m,n] = (1/16) * sum_c f1[b,c,h,m] * f2[b,c,h,n]
// Block = 128 threads (4 waves, wave32). Block tile 64x64, wave tile 16x64.
// Staging: branch-free float4 global loads (W1 % 4 == 0, rows 16B aligned),
// packed (k,k+1) b32 LDS stores, register double-buffered across K-chunks.
// ---------------------------------------------------------------------------
__global__ __launch_bounds__(128) void corr_wmma_kernel(
    const float* __restrict__ f1, const float* __restrict__ f2,
    float* __restrict__ corr0)
{
    const int bh = blockIdx.y;              // 0 .. B*H-1
    const int b  = bh / HH;
    const int h  = bh - b * HH;
    const int mt = blockIdx.x % 3;
    const int nt = blockIdx.x / 3;
    const int m0 = mt * 64;
    const int n0 = nt * 64;
    const int tid  = threadIdx.x;
    const int lane = tid & 31;
    const int wave = tid >> 5;

    // LDS tiles: [spatial][k], row stride 40 halves = 80B (16B-aligned rows)
    __shared__ __align__(16) _Float16 At[64][40];
    __shared__ __align__(16) _Float16 Bt[64][40];

    v8f acc[4] = {};

    const long cst = (long)HH * WW1;        // channel stride (floats)
    const float* f1base = f1 + ((long)b * CCH * HH + h) * (long)WW1;
    const float* f2base = f2 + ((long)b * CCH * HH + h) * (long)WW2;

    // Staging task decomposition: thread -> (m4, kp).
    // Covers m = 4*m4 + j (j=0..3), k = 2*(kp + 8*p) + q (p,q in {0,1}).
    const int m4 = tid & 15;                // 16 float4 positions along width
    const int kp = tid >> 4;                // 0..7
    const bool aok = (m0 + 4 * m4) < WW1;   // whole float4 in/out (184 % 4 == 0)
    const bool bok = (n0 + 4 * m4) < WW2;
    const float* pa = f1base + (long)(2 * kp) * cst + (aok ? (m0 + 4 * m4) : 0);
    const float* pb = f2base + (long)(2 * kp) * cst + (bok ? (n0 + 4 * m4) : 0);

    float4 ra[2][2], rb[2][2];
    // prologue: load K-chunk 0
    #pragma unroll
    for (int p = 0; p < 2; ++p) {
        #pragma unroll
        for (int q = 0; q < 2; ++q) {
            const long o = (long)(16 * p + q) * cst;
            ra[p][q] = *(const float4*)(pa + o);
            rb[p][q] = *(const float4*)(pb + o);
        }
    }

    for (int c = 0; c < CCH / 32; ++c) {
        __syncthreads();                    // prior fragment reads complete
        // ---- stage registers -> LDS (pack two k-halves per b32) ----
        #pragma unroll
        for (int p = 0; p < 2; ++p) {
            const int k = 2 * (kp + 8 * p);
            const float* fa0 = (const float*)&ra[p][0];
            const float* fa1 = (const float*)&ra[p][1];
            const float* fb0 = (const float*)&rb[p][0];
            const float* fb1 = (const float*)&rb[p][1];
            #pragma unroll
            for (int j = 0; j < 4; ++j) {
                PackU ua, ub;
                ua.h[0] = (_Float16)fa0[j];
                ua.h[1] = (_Float16)fa1[j];
                ub.h[0] = (_Float16)fb0[j];
                ub.h[1] = (_Float16)fb1[j];
                *(unsigned int*)&At[4 * m4 + j][k] = aok ? ua.u : 0u;
                *(unsigned int*)&Bt[4 * m4 + j][k] = bok ? ub.u : 0u;
            }
        }
        __syncthreads();
        // ---- prefetch next chunk (overlaps with DS reads + WMMA) ----
        if (c + 1 < CCH / 32) {
            const long base = (long)(c + 1) * 32 * cst;
            #pragma unroll
            for (int p = 0; p < 2; ++p) {
                #pragma unroll
                for (int q = 0; q < 2; ++q) {
                    const long o = base + (long)(16 * p + q) * cst;
                    ra[p][q] = *(const float4*)(pa + o);
                    rb[p][q] = *(const float4*)(pb + o);
                }
            }
        }
        // ---- A fragment (ISA 7.12.2): lane L<16: M=L, K {0..7}u{16..23};
        //      lane L>=16: same M, K {8..15}u{24..31}
        FragU a;
        {
            const int mloc = wave * 16 + (lane & 15);
            const int koff = (lane < 16) ? 0 : 8;
            const v4u* qa = (const v4u*)&At[mloc][koff];   // 16B aligned
            a.u[0] = qa[0];                 // halves koff .. koff+7
            a.u[1] = qa[2];                 // halves koff+16 .. koff+23
        }
        // ---- B fragments + WMMA ----
        #pragma unroll
        for (int t = 0; t < 4; ++t) {
            FragU bf;
            const int nloc = t * 16 + (lane & 15);
            const int kh   = (lane < 16) ? 0 : 8;          // K {kh..kh+7}u{kh+16..kh+23}
            const v4u* qb = (const v4u*)&Bt[nloc][kh];     // 16B aligned
            bf.u[0] = qb[0];
            bf.u[1] = qb[2];
            acc[t] = __builtin_amdgcn_wmma_f32_16x16x32_f16(
                /*neg_a=*/false, a.h, /*neg_b=*/false, bf.h,
                /*c_mod=*/(short)0, acc[t], /*reuse_a=*/false, /*reuse_b=*/false);
        }
    }

    // ---- store: C/D layout: VGPR j -> M = j (lanes 0-15) / 8+j (lanes 16-31),
    //      N = lane & 15. Scale by 1/sqrt(256) = 1/16.
    const float scale = 0.0625f;
    const long rowbase = (long)bh * WW1;
    const int mrow0 = m0 + wave * 16 + ((lane < 16) ? 0 : 8);
    #pragma unroll
    for (int t = 0; t < 4; ++t) {
        const int n_glob = n0 + t * 16 + (lane & 15);
        if (n_glob >= WW2) continue;
        #pragma unroll
        for (int j = 0; j < 8; ++j) {
            const int m_glob = mrow0 + j;
            if (m_glob < WW1)
                corr0[(rowbase + m_glob) * (long)WW2 + n_glob] = acc[t][j] * scale;
        }
    }
}

// ---------------------------------------------------------------------------
// Kernel 2: width avg-pool x2 (exact: 184 -> 92 -> 46 -> 23)
// ---------------------------------------------------------------------------
__global__ void pool2_kernel(const float* __restrict__ in, float* __restrict__ out,
                             int Win, int Wout, int total)
{
    int i = blockIdx.x * blockDim.x + threadIdx.x;
    if (i >= total) return;
    int r = i / Wout;
    int w = i - r * Wout;
    float2 v = *(const float2*)(in + (long)r * Win + 2 * w);   // rows 8B aligned
    out[i] = 0.5f * (v.x + v.y);
}

// ---------------------------------------------------------------------------
// Kernel 3: Gaussian-spread linear sampling with zero padding, all 4 levels.
// One thread per (n, g, s); out channel = l*18 + g*9 + s.
// ---------------------------------------------------------------------------
__global__ void sample_kernel(const float* __restrict__ l0, const float* __restrict__ l1,
                              const float* __restrict__ l2, const float* __restrict__ l3,
                              const float* __restrict__ coords, const float* __restrict__ sigma,
                              float* __restrict__ out, int total)
{
    int i = blockIdx.x * blockDim.x + threadIdx.x;   // total = N * G * S
    if (i >= total) return;
    const int gs = i % (NG * NS);
    const int n  = i / (NG * NS);
    const int g  = gs / NS;
    const int s  = gs - g * NS;
    const int b  = n / (HH * WW1);
    const int hw = n - b * HH * WW1;
    const int h  = hw / WW1;
    const int w1 = hw - h * WW1;

    const long ci = (((long)b * NG + g) * HH + h) * (long)WW1 + w1;
    const float mu = coords[ci];
    const float sg = sigma[ci];
    const float x  = (float)(s - NS / 2) * sg + mu;

    const float* levs[NL] = { l0, l1, l2, l3 };
    const int    wid[NL]  = { WW2, WW2 / 2, WW2 / 4, WW2 / 8 };

    #pragma unroll
    for (int l = 0; l < NL; ++l) {
        const float xi = x / (float)(1 << l);
        const float x0 = floorf(xi);
        const float wr = xi - x0;           // weight of right neighbor
        const int i0 = (int)x0;
        const int i1 = i0 + 1;
        const int W  = wid[l];
        const float* row = levs[l] + (long)n * W;
        const float v0 = (i0 >= 0 && i0 < W) ? row[i0] : 0.0f;
        const float v1 = (i1 >= 0 && i1 < W) ? row[i1] : 0.0f;
        const float v = (1.0f - wr) * v0 + wr * v1;
        const long oi = (((long)b * (NL * NG * NS) + (l * NG * NS + g * NS + s)) * HH + h)
                        * (long)WW1 + w1;
        out[oi] = v;
    }
}

// ---------------------------------------------------------------------------
extern "C" void kernel_launch(void* const* d_in, const int* in_sizes, int n_in,
                              void* d_out, int out_size, void* d_ws, size_t ws_size,
                              hipStream_t stream) {
    const float* f1     = (const float*)d_in[0];
    const float* f2     = (const float*)d_in[1];
    const float* coords = (const float*)d_in[2];
    const float* sigma  = (const float*)d_in[3];
    float* out = (float*)d_out;
    float* ws  = (float*)d_ws;

    const int N = BB * HH * WW1;             // 58880 rows
    float* l0 = ws;                          // N * 184
    float* l1 = l0 + (long)N * WW2;          // N * 92
    float* l2 = l1 + (long)N * (WW2 / 2);    // N * 46
    float* l3 = l2 + (long)N * (WW2 / 4);    // N * 23

    // 1) correlation GEMMs (f16 WMMA, fp32 accumulate)
    dim3 g1(9, BB * HH);                     // 3x3 (M,N) tiles of 64 per (b,h)
    corr_wmma_kernel<<<g1, 128, 0, stream>>>(f1, f2, l0);

    // 2) pyramid
    {
        int t1 = N * (WW2 / 2);
        pool2_kernel<<<(t1 + 255) / 256, 256, 0, stream>>>(l0, l1, WW2, WW2 / 2, t1);
        int t2 = N * (WW2 / 4);
        pool2_kernel<<<(t2 + 255) / 256, 256, 0, stream>>>(l1, l2, WW2 / 2, WW2 / 4, t2);
        int t3 = N * (WW2 / 8);
        pool2_kernel<<<(t3 + 255) / 256, 256, 0, stream>>>(l2, l3, WW2 / 4, WW2 / 8, t3);
    }

    // 3) sampling
    {
        int ts = N * NG * NS;
        sample_kernel<<<(ts + 255) / 256, 256, 0, stream>>>(l0, l1, l2, l3,
                                                            coords, sigma, out, ts);
    }
}
